// MMTrackerMotionModule_44255343018947
// MI455X (gfx1250) — matrix operand order repaired and checked by reading, not compiled, not run on hardware
//
#include <hip/hip_runtime.h>
#include <hip/hip_bf16.h>

// ---------------------------------------------------------------------------
// MMTracker motion module for MI455X (gfx1250, wave32, WMMA).
//   corr = max over 9x9 shifts of (cur * shifted(prev))      [B,512,112,112]
//   h1 = relu(conv3x3(corr, w1,b1))  512->256
//   h2 = relu(conv3x3(h1,  w2,b2))   256->128
//   mv =      conv3x3(h2,  w3,b3)    128->2   (fp32 output)
// Convs: implicit GEMM on v_wmma_f32_16x16x32_bf16, fp32 accumulate.
// Block tile 64(M) x 128(N), 8 waves as 2(M) x 4(N), each wave a 2x2 grid of
// 16x16 accumulators (4 WMMAs per K-substep, ds:wmma = 2:1).  K ordered
// k' = kidx*CIN + cin (one (dy,dx) per 64-wide K-step -> division-free,
// branchless im2col staging).  Double-buffered LDS, async global->LDS
// weight staging, single barrier per stage.
// ---------------------------------------------------------------------------

typedef __bf16 v16bf __attribute__((ext_vector_type(16)));
typedef __bf16 v8bf  __attribute__((ext_vector_type(8)));
typedef float  v8f   __attribute__((ext_vector_type(8)));
typedef int    intv4 __attribute__((ext_vector_type(4)));

#define HH 112
#define WW 112
#define HWSZ (HH * WW)          // 12544
#define NBATCH 4
#define NTOT (NBATCH * HWSZ)    // 50176 spatial positions
#define PADR 4                  // 9x9 window
#define LDSPITCH 72             // 64 halves + 8 pad (breaks bank-conflict stride)

#if __has_builtin(__builtin_amdgcn_global_load_async_to_lds_b128) && \
    __has_builtin(__builtin_amdgcn_s_wait_asynccnt)
#define HAVE_ASYNC_LDS 1
#else
#define HAVE_ASYNC_LDS 0
#endif

#if HAVE_ASYNC_LDS
// b128 async copy: global (AS1 int4*) -> LDS (AS3 int4*)
__device__ __forceinline__ void async_copy_b128(const __bf16* gsrc, __bf16* ldst) {
    __attribute__((address_space(1))) intv4* g =
        (__attribute__((address_space(1))) intv4*)(intv4*)(void*)const_cast<__bf16*>(gsrc);
    __attribute__((address_space(3))) intv4* l =
        (__attribute__((address_space(3))) intv4*)(intv4*)(void*)ldst;
    __builtin_amdgcn_global_load_async_to_lds_b128(g, l, 0, 0);
}
#endif

// ---------------------------------------------------------------------------
// 1) fp32 -> bf16 weight conversion with K-permutation:
//    dst[cout][kidx*CIN + cin] = (bf16) src[cout][cin*9 + kidx]
// ---------------------------------------------------------------------------
template <int CIN>
__global__ __launch_bounds__(256) void wcvt_perm_kernel(const float* __restrict__ src,
                                                        __bf16* __restrict__ dst, int n) {
    int i = blockIdx.x * 256 + threadIdx.x;
    if (i < n) {
        const int ktot = 9 * CIN;
        int cout = i / ktot;
        int r = i - cout * ktot;
        int kidx = r / CIN;            // CIN is a power of two -> shift
        int cin = r & (CIN - 1);
        dst[i] = (__bf16)src[((size_t)cout * CIN + cin) * 9 + kidx];
    }
}

// ---------------------------------------------------------------------------
// 2) correlation-max: per-channel 9x9 shifted product, max aggregated.
//    One block = one 16x16 spatial tile of one (b,c) plane; 24x24 halo in LDS.
// ---------------------------------------------------------------------------
__global__ __launch_bounds__(256) void corr_max_kernel(const float* __restrict__ cur,
                                                       const float* __restrict__ prev,
                                                       __bf16* __restrict__ out) {
    __shared__ float patch[24][24];

    const int plane = blockIdx.z;                 // b*512 + c
    const int ty0 = blockIdx.y * 16;
    const int tx0 = blockIdx.x * 16;
    const int tid = threadIdx.y * 16 + threadIdx.x;

    const float* prev_p = prev + (size_t)plane * HWSZ;
    const float* cur_p  = cur  + (size_t)plane * HWSZ;

#pragma unroll
    for (int rep = 0; rep < 3; ++rep) {           // 3*256 >= 24*24
        int i = rep * 256 + tid;
        if (i < 24 * 24) {
            int ly = i / 24, lx = i - ly * 24;
            int gy = ty0 + ly - PADR;
            int gx = tx0 + lx - PADR;
            bool ok = ((unsigned)gy < HH) & ((unsigned)gx < WW);
            int cy = ok ? gy : 0;
            int cx = ok ? gx : 0;
            float raw = prev_p[cy * WW + cx];     // unconditional load
            patch[ly][lx] = ok ? raw : 0.0f;      // v_cndmask
        }
    }
    __syncthreads();

    const int y = ty0 + threadIdx.y;
    const int x = tx0 + threadIdx.x;
    const float c = cur_p[y * WW + x];

    float m = -__builtin_inff();
#pragma unroll
    for (int dy = 0; dy < 9; ++dy) {
#pragma unroll
        for (int dx = 0; dx < 9; ++dx) {
            m = fmaxf(m, c * patch[threadIdx.y + dy][threadIdx.x + dx]);
        }
    }
    out[(size_t)plane * HWSZ + y * WW + x] = (__bf16)m;
}

// ---------------------------------------------------------------------------
// 3) conv3x3 as implicit GEMM with WMMA bf16 -> fp32.
// ---------------------------------------------------------------------------
template <int CIN, int COUT, bool RELU, bool OUT_F32>
__global__ __launch_bounds__(256) void conv3x3_wmma_kernel(
    const __bf16* __restrict__ act_in,   // [B][CIN][112][112]
    const __bf16* __restrict__ wbf,      // [COUT][9*CIN] (kidx-major K)
    const float* __restrict__ bias,      // [COUT]
    __bf16* __restrict__ act_out,        // [B][COUT][112][112] (if !OUT_F32)
    float* __restrict__ out_f32) {       // same shape fp32    (if OUT_F32)

    constexpr int KTOT = CIN * 9;
    constexpr int KSTEPS = KTOT / 64;
    constexpr bool FULL_M = (COUT % 64) == 0;
    static_assert((CIN % 64) == 0, "K-step must stay within one kidx");

    __shared__ __attribute__((aligned(16))) __bf16 As[2][64][LDSPITCH];   // [m][k]
    __shared__ __attribute__((aligned(16))) __bf16 Bs[2][128][LDSPITCH];  // [n][k]

    const int tid = threadIdx.x;
    const int lane = tid & 31;
    const int wave = tid >> 5;       // 0..7
    const int waveM = wave & 1;      // 2 tiles of 32 in M
    const int waveN = wave >> 1;     // 4 tiles of 32 in N

    const int ntile = blockIdx.x;    // 0..391
    const int mtile = blockIdx.y;
    const int n0 = ntile * 128;
    const int m0 = mtile * 64;

    // ---- B staging: thread owns n_local = tid&127 (adjacent threads ->
    //      adjacent x, coalesced), 32 consecutive cin's within the step.
    const int bn_local = tid & 127;
    const int bkg = (tid >> 7) * 32;        // 0 or 32
    const int n_glob = n0 + bn_local;
    const int bb = n_glob / HWSZ;
    const int brem = n_glob - bb * HWSZ;
    const int by = brem / WW;
    const int bx = brem - by * WW;
    const __bf16* act_base = act_in + (size_t)bb * CIN * HWSZ;

    // ---- A staging: thread owns row tid>>2, 16 consecutive k's.
    const int am_row = tid >> 2;
    const int ak0 = (tid & 3) * 16;

    const int lhalf = lane & 15;
    const int lhi = lane >> 4;

    v8f acc[2][2] = {{{}, {}}, {{}, {}}};

    auto stage = [&](int ks, int buf) {
        const int kbase = ks * 64;
        // ---- A (weights): contiguous 32B per thread ----
        if constexpr (FULL_M) {
            const __bf16* src = wbf + (size_t)(m0 + am_row) * KTOT + kbase + ak0;
            __bf16* dst = &As[buf][am_row][ak0];
#if HAVE_ASYNC_LDS
            async_copy_b128(src, dst);
            async_copy_b128(src + 8, dst + 8);
#else
            v8bf q0 = *(const v8bf*)src;
            v8bf q1 = *(const v8bf*)(src + 8);
            *(v8bf*)dst       = q0;
            *(v8bf*)(dst + 8) = q1;
#endif
        } else {
            const int row = m0 + am_row;
            const bool ok = row < COUT;
            const __bf16* src = wbf + (size_t)(ok ? row : 0) * KTOT + kbase + ak0;
            v8bf q0 = *(const v8bf*)src;          // unconditional (row 0 safe)
            v8bf q1 = *(const v8bf*)(src + 8);
            v8bf z = {};
            *(v8bf*)&As[buf][am_row][ak0]     = ok ? q0 : z;
            *(v8bf*)&As[buf][am_row][ak0 + 8] = ok ? q1 : z;
        }
        // ---- B (im2col): one kidx per step -> one (dy,dx), one predicate,
        //      32 loads at stride HWSZ via immediate offsets. ----
        const int kidx = kbase / CIN;                 // shift (CIN pow2), uniform
        const int t3 = kidx / 3;                      // uniform, once per stage
        const int dy = t3 - 1;
        const int dx = kidx - t3 * 3 - 1;
        const int cin0 = (kbase & (CIN - 1)) + bkg;
        const int yy = by + dy;
        const int xx = bx + dx;
        const bool ok = ((unsigned)yy < HH) & ((unsigned)xx < WW);
        const int cy = ok ? yy : 0;
        const int cx = ok ? xx : 0;
        const __bf16* p = act_base + (size_t)cin0 * HWSZ + cy * WW + cx;
        union { __bf16 h[32]; v8bf v[4]; } pk;
#pragma unroll
        for (int i = 0; i < 32; ++i) {
            __bf16 raw = p[i * HWSZ];                 // imm-offset loads
            pk.h[i] = ok ? raw : (__bf16)0.0f;
        }
#pragma unroll
        for (int j = 0; j < 4; ++j)
            *(v8bf*)&Bs[buf][bn_local][bkg + j * 8] = pk.v[j];
    };

    stage(0, 0);
#if HAVE_ASYNC_LDS
    if constexpr (FULL_M) __builtin_amdgcn_s_wait_asynccnt(0);
#endif
    __syncthreads();

    for (int ks = 0; ks < KSTEPS; ++ks) {
        const int buf = ks & 1;
        if (ks + 1 < KSTEPS) stage(ks + 1, buf ^ 1);

#pragma unroll
        for (int s = 0; s < 2; ++s) {
            v16bf afrag[2], bfrag[2];
            const int ka0 = s * 32 + (lhi ? 8 : 0);
            const int kb0 = s * 32 + (lhi ? 8 : 0);
#pragma unroll
            for (int mi = 0; mi < 2; ++mi) {
                const int row = waveM * 32 + mi * 16 + lhalf;
                v8bf a_lo = *(const v8bf*)&As[buf][row][ka0];
                v8bf a_hi = *(const v8bf*)&As[buf][row][ka0 + 16];
                afrag[mi] = __builtin_shufflevector(a_lo, a_hi, 0, 1, 2, 3, 4, 5, 6, 7,
                                                    8, 9, 10, 11, 12, 13, 14, 15);
            }
#pragma unroll
            for (int ni = 0; ni < 2; ++ni) {
                const int row = waveN * 32 + ni * 16 + lhalf;
                v8bf b_lo = *(const v8bf*)&Bs[buf][row][kb0];
                v8bf b_hi = *(const v8bf*)&Bs[buf][row][kb0 + 16];
                bfrag[ni] = __builtin_shufflevector(b_lo, b_hi, 0, 1, 2, 3, 4, 5, 6, 7,
                                                    8, 9, 10, 11, 12, 13, 14, 15);
            }
#pragma unroll
            for (int mi = 0; mi < 2; ++mi)
#pragma unroll
                for (int ni = 0; ni < 2; ++ni)
                    acc[mi][ni] = __builtin_amdgcn_wmma_f32_16x16x32_bf16(
                        false, afrag[mi], false, bfrag[ni], (short)0, acc[mi][ni],
                        false, false);
        }
#if HAVE_ASYNC_LDS
        if constexpr (FULL_M) __builtin_amdgcn_s_wait_asynccnt(0);
#endif
        __syncthreads();
    }

    // ---- epilogue: bias (+ReLU), scatter to NCHW ----
#pragma unroll
    for (int ni = 0; ni < 2; ++ni) {
        const int sn = n0 + waveN * 32 + ni * 16 + lhalf;
        const int sb = sn / HWSZ;
        const int srem = sn - sb * HWSZ;
        const int sy = srem / WW;
        const int sx = srem - sy * WW;
#pragma unroll
        for (int mi = 0; mi < 2; ++mi) {
#pragma unroll
            for (int v = 0; v < 8; ++v) {
                const int cout = m0 + waveM * 32 + mi * 16 + v + lhi * 8;
                if (cout < COUT) {
                    float r = acc[mi][ni][v] + bias[cout];
                    if (RELU) r = fmaxf(r, 0.0f);
                    const size_t oidx = (((size_t)sb * COUT + cout) * HH + sy) * WW + sx;
                    if constexpr (OUT_F32) out_f32[oidx] = r;
                    else                   act_out[oidx] = (__bf16)r;
                }
            }
        }
    }
}

// ---------------------------------------------------------------------------
// Host launcher
// ---------------------------------------------------------------------------
extern "C" void kernel_launch(void* const* d_in, const int* in_sizes, int n_in,
                              void* d_out, int out_size, void* d_ws, size_t ws_size,
                              hipStream_t stream) {
    const float* feat_cur  = (const float*)d_in[0];
    const float* feat_prev = (const float*)d_in[1];
    const float* w1 = (const float*)d_in[2];
    const float* b1 = (const float*)d_in[3];
    const float* w2 = (const float*)d_in[4];
    const float* b2 = (const float*)d_in[5];
    const float* w3 = (const float*)d_in[6];
    const float* b3 = (const float*)d_in[7];

    char* ws = (char*)d_ws;
    size_t off = 0;
    auto carve = [&](size_t bytes) -> char* {
        char* p = ws + off;
        off = (off + bytes + 255) & ~(size_t)255;
        return p;
    };
    __bf16* corr_bf = (__bf16*)carve((size_t)NBATCH * 512 * HWSZ * 2);
    __bf16* h1      = (__bf16*)carve((size_t)NBATCH * 256 * HWSZ * 2);
    __bf16* h2      = (__bf16*)carve((size_t)NBATCH * 128 * HWSZ * 2);
    __bf16* w1bf    = (__bf16*)carve((size_t)256 * 512 * 9 * 2);
    __bf16* w2bf    = (__bf16*)carve((size_t)128 * 256 * 9 * 2);
    __bf16* w3bf    = (__bf16*)carve((size_t)2 * 128 * 9 * 2);

    {
        int n1 = 256 * 512 * 9, n2 = 128 * 256 * 9, n3 = 2 * 128 * 9;
        wcvt_perm_kernel<512><<<(n1 + 255) / 256, 256, 0, stream>>>(w1, w1bf, n1);
        wcvt_perm_kernel<256><<<(n2 + 255) / 256, 256, 0, stream>>>(w2, w2bf, n2);
        wcvt_perm_kernel<128><<<(n3 + 255) / 256, 256, 0, stream>>>(w3, w3bf, n3);
    }
    {
        dim3 grid(WW / 16, HH / 16, NBATCH * 512);
        dim3 block(16, 16);
        corr_max_kernel<<<grid, block, 0, stream>>>(feat_cur, feat_prev, corr_bf);
    }

    const int ntiles = NTOT / 128;  // 392
    conv3x3_wmma_kernel<512, 256, true, false>
        <<<dim3(ntiles, 256 / 64), 256, 0, stream>>>(corr_bf, w1bf, b1, h1, nullptr);
    conv3x3_wmma_kernel<256, 128, true, false>
        <<<dim3(ntiles, 128 / 64), 256, 0, stream>>>(h1, w2bf, b2, h2, nullptr);
    conv3x3_wmma_kernel<128, 2, false, true>
        <<<dim3(ntiles, 1), 256, 0, stream>>>(h2, w3bf, b3, nullptr, (float*)d_out);
}
